// GraspEvaluator_67577015435423
// MI455X (gfx1250) — matrix operand order repaired and checked by reading, not compiled, not run on hardware
//
#include <hip/hip_runtime.h>
#include <hip/hip_bf16.h>
#include <cstddef>

typedef _Float16 half_t;
typedef __attribute__((ext_vector_type(16))) _Float16 v16h;
typedef __attribute__((ext_vector_type(8)))  float    v8f;

#define EPSV 1e-5f

// ---------------------------------------------------------------------------
// WMMA fragment loaders (layouts per CDNA5 ISA 7.12.2, wave32)
// A 16x32 f16: lanes 0-15 hold M=0..15, K = {0..7,16..23}; lanes 16-31 K = {8..15,24..31}
// B 32x16 f16: lane L holds col N=L&15, K same as A: lanes 0-15 {0..7,16..23}; lanes 16-31 {8..15,24..31}
// C/D 16x16 f32: lane L col N=L&15, VGPR r row M = r + (L>=16?8:0)
// ---------------------------------------------------------------------------
__device__ inline v16h load_a_frag(const half_t* __restrict__ row, int k0, int hi) {
  const half_t* p = row + k0 + hi * 8;
  v16h a;
#pragma unroll
  for (int i = 0; i < 8; ++i) { a[i] = p[i]; a[8 + i] = p[16 + i]; }
  return a;
}
__device__ inline v16h load_b_frag(const half_t* __restrict__ wcol, int k0, int hi) {
  const half_t* p = wcol + k0 + hi * 8;
  v16h b;
#pragma unroll
  for (int i = 0; i < 8; ++i) { b[i] = p[i]; b[8 + i] = p[16 + i]; }
  return b;
}

struct LayerP {
  const half_t* wt;   // [Cout][Kpad] f16 (transposed, K zero-padded)
  const float* b;
  const float* be;
  const float* g;
  const float* m;
  const float* v;
};

// One dense layer: bufIn [NS][LSIN] f16 (logical K = KIN) -> bufOut [NS][LSOUT] f16,
// y = relu(bn(x @ W + b)).  8 waves, each wave owns 16x16 output tiles.
template<int NS, int KIN, int LSIN, int COUT, int LSOUT>
__device__ inline void mlp_layer(const half_t* __restrict__ bufIn,
                                 half_t* __restrict__ bufOut,
                                 LayerP P, int tid) {
  const int lane = tid & 31, wave = tid >> 5;
  const int hi = (lane >> 4) & 1, ln = lane & 15;
  const int MT = NS / 16, NT = COUT / 16, T = MT * NT;
  for (int t = wave; t < T; t += 8) {
    const int mt = t % MT, nt = t / MT;
    const half_t* arow = bufIn + (mt * 16 + ln) * LSIN;
    const half_t* wcol = P.wt + (size_t)(nt * 16 + ln) * KIN;
    v8f acc = {};
    for (int k0 = 0; k0 < KIN; k0 += 32) {
      v16h af = load_a_frag(arow, k0, hi);
      v16h bf = load_b_frag(wcol, k0, hi);
      acc = __builtin_amdgcn_wmma_f32_16x16x32_f16(false, af, false, bf,
                                                   (short)0, acc, false, false);
    }
    const int col = nt * 16 + ln;
    const float sc = P.g[col] * rsqrtf(P.v[col] + EPSV);
    const float sh = P.be[col] - P.m[col] * sc;
    const float bb = P.b[col];
#pragma unroll
    for (int r = 0; r < 8; ++r) {
      float val = (acc[r] + bb) * sc + sh;
      val = val > 0.f ? val : 0.f;
      bufOut[(mt * 16 + hi * 8 + r) * LSOUT + col] = (half_t)val;
    }
  }
}

// ---------------------------------------------------------------------------
// Fused SA stage: ball-query -> gather+center -> MLP(3 layers, WMMA) -> max-pool.
// One workgroup (256 thr / 8 waves) per (batch, centroid).
// MODE 0: feats synthesized from xyz + pc/gripper label (SA1)
// MODE 1: feats gathered from f16 feature map (SA2)
// MODE 2: group-all, no centering (SA3)
// ---------------------------------------------------------------------------
template<int MODE, int NS, int K0, int C1, int C2, int C3, int NSRC, int FIN>
__global__ __launch_bounds__(256)
void sa_kernel(const float* __restrict__ xyz,
               const half_t* __restrict__ feats_in,
               const float* __restrict__ new_xyz,
               float r2, int npoint,
               LayerP L1, LayerP L2, LayerP L3,
               half_t* __restrict__ feats_out) {
  constexpr int LSA = (K0 > C2) ? K0 : C2;
  constexpr int LSB = (C1 > C3) ? C1 : C3;
  extern __shared__ char smem[];
  half_t* bufA = (half_t*)smem;
  half_t* bufB = bufA + NS * LSA;
  int* sidx = (int*)(bufB + NS * LSB);
  int* scnt = sidx + NS;

  const int bc = blockIdx.x;
  const int b = bc / npoint;
  const int tid = threadIdx.x;
  const float* X = xyz + (size_t)b * NSRC * 3;

  float cx = 0.f, cy = 0.f, cz = 0.f;
  if (MODE < 2) {
    const float* nx = new_xyz + (size_t)bc * 3;
    cx = nx[0]; cy = nx[1]; cz = nx[2];
  }

  if (MODE < 2) {
    if (tid < 32) {  // wave0: ordered in-radius index selection (matches sorted ball_query)
      int count = 0;
      for (int base = 0; base < NSRC && count < NS; base += 32) {
        int i = base + tid;
        bool ok = false;
        if (i < NSRC) {
          float dx = X[i * 3] - cx, dy = X[i * 3 + 1] - cy, dz = X[i * 3 + 2] - cz;
          ok = (dx * dx + dy * dy + dz * dz) <= r2;
        }
        unsigned mask = __builtin_amdgcn_ballot_w32(ok);
        int pos = count + __popc(mask & ((1u << tid) - 1u));
        if (ok && pos < NS) sidx[pos] = i;
        count += __popc(mask);
      }
      if (count > NS) count = NS;
      if (tid == 0) {
        if (count == 0) { sidx[0] = NSRC - 1; count = 1; }
        *scnt = count;
      }
    }
    __syncthreads();
    {
      int cnt = *scnt;
      int pad = sidx[0];
      for (int p = cnt + tid; p < NS; p += 256) sidx[p] = pad;
    }
    __syncthreads();
  } else {
    for (int p = tid; p < NS; p += 256) sidx[p] = p;
    __syncthreads();
  }

  // Build grouped input tile [NS][K0] f16 (zero K-pad)
  for (int e = tid; e < NS * K0; e += 256) {
    int s = e / K0, ch = e - s * K0;
    int idx = sidx[s];
    float val = 0.f;
    if (ch < 3) {
      float p = X[idx * 3 + ch];
      if (MODE < 2) p -= (ch == 0 ? cx : (ch == 1 ? cy : cz));
      val = p;
    } else if (MODE == 0) {
      if (ch < 6) val = X[idx * 3 + (ch - 3)];
      else if (ch == 6) val = (idx < 1024) ? 1.f : 0.f;  // pc-vs-gripper label
    } else {
      int f = ch - 3;
      if (f < FIN) val = (float)feats_in[((size_t)b * NSRC + idx) * FIN + f];
    }
    bufA[s * LSA + ch] = (half_t)val;
  }
  __syncthreads();

  mlp_layer<NS, K0, LSA, C1, LSB>(bufA, bufB, L1, tid);
  __syncthreads();
  mlp_layer<NS, C1, LSB, C2, LSA>(bufB, bufA, L2, tid);
  __syncthreads();
  mlp_layer<NS, C2, LSA, C3, LSB>(bufA, bufB, L3, tid);
  __syncthreads();

  for (int ch = tid; ch < C3; ch += 256) {  // max over samples
    float mx = -3.0e38f;
    for (int s = 0; s < NS; ++s) {
      float vv = (float)bufB[s * LSB + ch];
      mx = vv > mx ? vv : mx;
    }
    feats_out[(size_t)bc * C3 + ch] = (half_t)mx;
  }
}

constexpr size_t sa_smem(int NS, int LSA, int LSB) {
  return (size_t)NS * LSA * 2 + (size_t)NS * LSB * 2 + (size_t)NS * 4 + 16;
}

// ---------------------------------------------------------------------------
// Build xyz[B][1030][3]: 1024 pc points + 6 quat-rotated gripper points
// ---------------------------------------------------------------------------
__global__ void build_xyz(const float* __restrict__ quat, const float* __restrict__ trans,
                          const float* __restrict__ pc, const float* __restrict__ cp,
                          float* __restrict__ xyz) {
  const int NTOT = 1030;
  int id = blockIdx.x * blockDim.x + threadIdx.x;
  if (id >= 256 * NTOT) return;
  int b = id / NTOT, n = id - b * NTOT;
  float px, py, pz;
  if (n < 1024) {
    const float* p = pc + ((size_t)b * 1024 + n) * 3;
    px = p[0]; py = p[1]; pz = p[2];
  } else {
    int g = n - 1024;
    const float* q = quat + b * 4;
    float qw = q[0], qx = q[1], qy = q[2], qz = q[3];
    float inv = rsqrtf(qw * qw + qx * qx + qy * qy + qz * qz);
    qw *= inv; qx *= inv; qy *= inv; qz *= inv;
    const float* c = cp + g * 3;
    float X = c[0], Y = c[1], Z = c[2];
    px = (1 - 2 * (qy * qy + qz * qz)) * X + 2 * (qx * qy - qw * qz) * Y + 2 * (qx * qz + qw * qy) * Z + trans[b * 3 + 0];
    py = 2 * (qx * qy + qw * qz) * X + (1 - 2 * (qx * qx + qz * qz)) * Y + 2 * (qy * qz - qw * qx) * Z + trans[b * 3 + 1];
    pz = 2 * (qx * qz - qw * qy) * X + 2 * (qy * qz + qw * qx) * Y + (1 - 2 * (qx * qx + qy * qy)) * Z + trans[b * 3 + 2];
  }
  float* o = xyz + (size_t)id * 3;
  o[0] = px; o[1] = py; o[2] = pz;
}

// ---------------------------------------------------------------------------
// Farthest point sampling: one workgroup per batch, sequential over npoint,
// parallel min-distance update + argmax tree-reduce (first-index tiebreak).
// Also gathers new_xyz.
// ---------------------------------------------------------------------------
#define FPS_NMAX 1030
__global__ __launch_bounds__(256)
void fps_kernel(const float* __restrict__ xyz, int n, int npoint,
                int* __restrict__ fidx, float* __restrict__ new_xyz) {
  __shared__ float sx[FPS_NMAX * 3];
  __shared__ float sd[FPS_NMAX];
  __shared__ float rv[256];
  __shared__ int ri[256];
  __shared__ int s_cur;
  int b = blockIdx.x, tid = threadIdx.x;
  const float* X = xyz + (size_t)b * n * 3;
  for (int i = tid; i < n; i += 256) {
    sx[i * 3] = X[i * 3]; sx[i * 3 + 1] = X[i * 3 + 1]; sx[i * 3 + 2] = X[i * 3 + 2];
    sd[i] = 1e10f;
  }
  if (tid == 0) s_cur = 0;
  __syncthreads();
  for (int it = 0; it < npoint; ++it) {
    int cur = s_cur;
    if (tid == 0) {
      fidx[b * npoint + it] = cur;
      new_xyz[(size_t)(b * npoint + it) * 3 + 0] = sx[cur * 3];
      new_xyz[(size_t)(b * npoint + it) * 3 + 1] = sx[cur * 3 + 1];
      new_xyz[(size_t)(b * npoint + it) * 3 + 2] = sx[cur * 3 + 2];
    }
    float cx = sx[cur * 3], cy = sx[cur * 3 + 1], cz = sx[cur * 3 + 2];
    float bv = -1.f; int bi = 0;
    for (int i = tid; i < n; i += 256) {
      float dx = sx[i * 3] - cx, dy = sx[i * 3 + 1] - cy, dz = sx[i * 3 + 2] - cz;
      float d = dx * dx + dy * dy + dz * dz;
      float nd = sd[i] < d ? sd[i] : d;
      sd[i] = nd;
      if (nd > bv) { bv = nd; bi = i; }
    }
    rv[tid] = bv; ri[tid] = bi;
    __syncthreads();
    for (int s = 128; s > 0; s >>= 1) {
      if (tid < s) {
        if (rv[tid + s] > rv[tid] || (rv[tid + s] == rv[tid] && ri[tid + s] < ri[tid])) {
          rv[tid] = rv[tid + s]; ri[tid] = ri[tid + s];
        }
      }
      __syncthreads();
    }
    if (tid == 0) s_cur = ri[0];
    __syncthreads();
  }
}

// ---------------------------------------------------------------------------
// Weight prep: f32 [Cin][Cout] -> f16 transposed [Cout][Kpad], K zero-padded
// ---------------------------------------------------------------------------
__global__ void prep_wt(const float* __restrict__ W, half_t* __restrict__ Wt,
                        int cin, int cout, int kpad) {
  int i = blockIdx.x * blockDim.x + threadIdx.x;
  if (i >= cout * kpad) return;
  int n = i / kpad, k = i - n * kpad;
  Wt[i] = (half_t)((k < cin) ? W[k * cout + n] : 0.f);
}

// ---------------------------------------------------------------------------
// FC GEMM: A f16 [M][K] @ Wt f16 [N][K]^T + bias -> BN -> ReLU -> f16 [M][N]
// ---------------------------------------------------------------------------
template<int M, int K>
__global__ __launch_bounds__(256)
void fc_kernel(const half_t* __restrict__ A, const half_t* __restrict__ Wt,
               const float* __restrict__ bias, const float* __restrict__ g,
               const float* __restrict__ be, const float* __restrict__ m,
               const float* __restrict__ v, half_t* __restrict__ out, int N) {
  const int tid = threadIdx.x;
  const int lane = tid & 31, wave = tid >> 5;
  const int hi = (lane >> 4) & 1, ln = lane & 15;
  const int MT = M / 16, NT = N / 16;
  int tile = blockIdx.x * 8 + wave;
  if (tile >= MT * NT) return;
  int mt = tile % MT, nt = tile / MT;
  const half_t* arow = A + (size_t)(mt * 16 + ln) * K;
  const half_t* wcol = Wt + (size_t)(nt * 16 + ln) * K;
  v8f acc = {};
  for (int k0 = 0; k0 < K; k0 += 32) {
    v16h af = load_a_frag(arow, k0, hi);
    v16h bf = load_b_frag(wcol, k0, hi);
    acc = __builtin_amdgcn_wmma_f32_16x16x32_f16(false, af, false, bf,
                                                 (short)0, acc, false, false);
  }
  const int col = nt * 16 + ln;
  const float sc = g[col] * rsqrtf(v[col] + EPSV);
  const float sh = be[col] - m[col] * sc;
  const float bb = bias[col];
#pragma unroll
  for (int r = 0; r < 8; ++r) {
    float val = (acc[r] + bb) * sc + sh;
    val = val > 0.f ? val : 0.f;
    out[(size_t)(mt * 16 + hi * 8 + r) * N + col] = (half_t)val;
  }
}

__global__ void cls_kernel(const half_t* __restrict__ x, const float* __restrict__ W,
                           const float* __restrict__ b, float* __restrict__ out) {
  int bt = threadIdx.x;  // one thread per batch row
  const half_t* xr = x + (size_t)bt * 1024;
  float s = 0.f;
  for (int j = 0; j < 1024; ++j) s += (float)xr[j] * W[j];
  out[bt] = s + b[0];
}

// ---------------------------------------------------------------------------
// Host orchestration.
// Input layout (jax tree-flatten: top-level dict insertion order, nested dicts
// alphabetical, lists in order):
//  0 quat  1 trans  2 pc  3 control_points
//  4-7  bn1 {be,g,m,v}   8-11 bn2 {be,g,m,v}
//  12 cls.W  13 cls.b    14 fc1.W  15 fc1.b   16 fc2.W  17 fc2.b
//  18+ sa[i][layer]: W, b, bn{be,g,m,v}  (6 leaves per layer, 9 layers)
// ---------------------------------------------------------------------------
extern "C" void kernel_launch(void* const* d_in, const int* in_sizes, int n_in,
                              void* d_out, int out_size, void* d_ws, size_t ws_size,
                              hipStream_t stream) {
  (void)in_sizes; (void)n_in; (void)out_size; (void)ws_size;
  auto F = [&](int i) { return (const float*)d_in[i]; };

  char* ws = (char*)d_ws;
  size_t off = 0;
  auto carve = [&](size_t bytes) -> char* {
    char* p = ws + off;
    off = (off + bytes + 255) & ~(size_t)255;
    return p;
  };
  float*  xyz0   = (float*)carve((size_t)256 * 1030 * 3 * 4);
  int*    fidx1  = (int*)  carve((size_t)256 * 128 * 4);
  float*  nxyz1  = (float*)carve((size_t)256 * 128 * 3 * 4);
  half_t* feats1 = (half_t*)carve((size_t)256 * 128 * 128 * 2);
  int*    fidx2  = (int*)  carve((size_t)256 * 32 * 4);
  float*  nxyz2  = (float*)carve((size_t)256 * 32 * 3 * 4);
  half_t* feats2 = (half_t*)carve((size_t)256 * 32 * 256 * 2);
  half_t* feats3 = (half_t*)carve((size_t)256 * 512 * 2);
  half_t* x1     = (half_t*)carve((size_t)256 * 1024 * 2);
  half_t* x2     = (half_t*)carve((size_t)256 * 1024 * 2);

  struct WSp { int src, cin, cout, kpad; };
  const WSp wspec[11] = {
    {18, 7, 64, 32},    {24, 64, 64, 64},    {30, 64, 128, 64},
    {36, 131, 128, 160},{42, 128, 128, 128}, {48, 128, 256, 128},
    {54, 259, 256, 288},{60, 256, 256, 256}, {66, 256, 512, 256},
    {14, 512, 1024, 512},{16, 1024, 1024, 1024}
  };
  half_t* wt[11];
  for (int i = 0; i < 11; ++i) {
    size_t elems = (size_t)wspec[i].cout * wspec[i].kpad;
    wt[i] = (half_t*)carve(elems * 2);
    prep_wt<<<(int)((elems + 255) / 256), 256, 0, stream>>>(
        F(wspec[i].src), wt[i], wspec[i].cin, wspec[i].cout, wspec[i].kpad);
  }

  auto LP = [&](int slot, int src) {
    LayerP p;
    p.wt = wt[slot];
    p.b = F(src + 1); p.be = F(src + 2); p.g = F(src + 3);
    p.m = F(src + 4); p.v = F(src + 5);
    return p;
  };

  build_xyz<<<(256 * 1030 + 255) / 256, 256, 0, stream>>>(F(0), F(1), F(2), F(3), xyz0);

  fps_kernel<<<256, 256, 0, stream>>>(xyz0, 1030, 128, fidx1, nxyz1);

  sa_kernel<0, 64, 32, 64, 64, 128, 1030, 4>
      <<<256 * 128, 256, sa_smem(64, 64, 128), stream>>>(
          xyz0, nullptr, nxyz1, 0.02f * 0.02f, 128,
          LP(0, 18), LP(1, 24), LP(2, 30), feats1);

  fps_kernel<<<256, 256, 0, stream>>>(nxyz1, 128, 32, fidx2, nxyz2);

  sa_kernel<1, 128, 160, 128, 128, 256, 128, 128>
      <<<256 * 32, 256, sa_smem(128, 160, 256), stream>>>(
          nxyz1, feats1, nxyz2, 0.04f * 0.04f, 32,
          LP(3, 36), LP(4, 42), LP(5, 48), feats2);

  sa_kernel<2, 32, 288, 256, 256, 512, 32, 256>
      <<<256, 256, sa_smem(32, 288, 512), stream>>>(
          nxyz2, feats2, nullptr, 0.f, 1,
          LP(6, 54), LP(7, 60), LP(8, 66), feats3);

  fc_kernel<256, 512><<<128, 256, 0, stream>>>(
      feats3, wt[9], F(15), F(5), F(4), F(6), F(7), x1, 1024);
  fc_kernel<256, 1024><<<128, 256, 0, stream>>>(
      x1, wt[10], F(17), F(9), F(8), F(10), F(11), x2, 1024);
  cls_kernel<<<1, 256, 0, stream>>>(x2, F(12), F(13), (float*)d_out);
}
